// WindowMHSA_19112604467888
// MI455X (gfx1250) — compile-verified, matrix operand order checked by reading, not run on hardware
//
#include <hip/hip_runtime.h>
#include <hip/hip_bf16.h>

typedef __attribute__((ext_vector_type(16))) __bf16 v16bf;
typedef __attribute__((ext_vector_type(8)))  float  v8f;

#define TPB    256
#define CCH    192
#define NHEADS 8
#define HD     24
#define WW     49
#define MPAD   64
#define XPITCH 200   // bf16 elements per padded X/AO row

// d_ws layout, in ushort (bf16) elements
#define WS_QKVW  0
#define WS_QKVB  (WS_QKVW + 576*192)      // 110592
#define WS_PROJW (WS_QKVB + 576)          // 111168
#define WS_PROJB (WS_PROJW + 192*192)     // 148032

// LDS byte offsets (dynamic shared)
#define L_XB   0                          // Xb: 64x200 bf16 (aliased by AO)
#define L_QS   25600                      // Qs: 8x64x32 bf16 (hd padded->32, zeroed)
#define L_KS   (25600 + 32768)            // Ks: 8x64x32 bf16
#define L_VT   (25600 + 65536)            // Vt: 8x32x64 bf16 (V transposed)
#define L_PS   (25600 + 98304)            // Ps: 8x64x64 bf16 (aliased by OutS f32 49x192)
#define L_TOTAL (25600 + 98304 + 65536)   // 189440 bytes

__device__ __forceinline__ unsigned short f2bf(float f) {
  unsigned int u = __builtin_bit_cast(unsigned int, f);
  u += 0x7FFFu + ((u >> 16) & 1u);        // round-to-nearest-even
  return (unsigned short)(u >> 16);
}
__device__ __forceinline__ float bf2f(unsigned short u) {
  unsigned int x = ((unsigned int)u) << 16;
  return __builtin_bit_cast(float, x);
}
__device__ __forceinline__ unsigned int f2bf_pk(float lo, float hi) {
  return (unsigned int)f2bf(lo) | ((unsigned int)f2bf(hi) << 16);
}

union Frag16 { v16bf v; unsigned int u[8]; };

// Load one 16x32 bf16 fragment (A or B role). `p` = per-lane row base (row
// chosen by caller from lane&15), K-pairs contiguous. ISA layout: lanes 0-15
// hold K={0..7,16..23}, lanes 16-31 hold K={8..15,24..31}.
__device__ __forceinline__ v16bf ld_frag(const unsigned short* p, int hf) {
  Frag16 f;
#pragma unroll
  for (int i = 0; i < 8; ++i) {
    int kb = 2*i + ((i >= 4) ? 8 : 0) + 8*hf;
    f.u[i] = *(const unsigned int*)(p + kb);
  }
  return f.v;
}

__device__ __forceinline__ v8f wmma_bf16(v16bf a, v16bf b, v8f c) {
  return __builtin_amdgcn_wmma_f32_16x16x32_bf16(false, a, false, b, (short)0, c,
                                                 false, false);
}

__global__ void prep_weights(const float* __restrict__ qkv_w,
                             const float* __restrict__ qkv_b,
                             const float* __restrict__ proj_w,
                             const float* __restrict__ proj_b,
                             unsigned short* __restrict__ ws) {
  int i = blockIdx.x * blockDim.x + threadIdx.x;
  if (i < 576*192) ws[WS_QKVW  + i] = f2bf(qkv_w[i]);
  if (i < 576)     ws[WS_QKVB  + i] = f2bf(qkv_b[i]);
  if (i < 192*192) ws[WS_PROJW + i] = f2bf(proj_w[i]);
  if (i < 192)     ws[WS_PROJB + i] = f2bf(proj_b[i]);
}

__global__ __launch_bounds__(TPB)
void win_mhsa(const float* __restrict__ x,
              const unsigned short* __restrict__ ws,
              float* __restrict__ out) {
  extern __shared__ __align__(16) unsigned char smem[];
  unsigned short* Xb = (unsigned short*)(smem + L_XB);
  unsigned short* AO = Xb;                            // alias: AO reuses Xb
  unsigned short* Qs = (unsigned short*)(smem + L_QS);
  unsigned short* Ks = (unsigned short*)(smem + L_KS);
  unsigned short* Vt = (unsigned short*)(smem + L_VT);
  unsigned short* Ps = (unsigned short*)(smem + L_PS);
  float*          OutS = (float*)(smem + L_PS);       // alias: OutS reuses Ps

  const int tid  = threadIdx.x;
  const int lane = tid & 31;
  const int wave = tid >> 5;
  const int hf   = lane >> 4;     // lane half selects K subset
  const int nn   = lane & 15;     // row (A) / column (B,C,D) within tile

  const unsigned short* gqkvW  = ws + WS_QKVW;
  const unsigned short* gqkvB  = ws + WS_QKVB;
  const unsigned short* gprojW = ws + WS_PROJW;
  const unsigned short* gprojB = ws + WS_PROJB;

  const int blk  = blockIdx.x;    // 4096 windows
  const int nIdx = blk >> 6;
  const int rem  = blk & 63;
  const int whi  = rem >> 3;
  const int wwi  = rem & 7;

  // ---- phase 0: zero padded Q/K, stage x window tile as bf16 ----
  for (int i = tid; i < NHEADS*64*32; i += TPB) { Qs[i] = 0; Ks[i] = 0; }
  for (int i = tid; i < CCH*WW; i += TPB) {
    int c = i / WW, p = i % WW;
    int wy = p / 7, wx = p % 7;
    float v = x[(((size_t)nIdx*CCH + c)*56 + (whi*7 + wy))*56 + (wwi*7 + wx)];
    Xb[p * XPITCH + c] = f2bf(v);
  }
  for (int i = tid; i < (MPAD - WW)*XPITCH; i += TPB) Xb[WW*XPITCH + i] = 0;
  __syncthreads();

  // ---- phase 1: QKV = Xb(64x192) @ qkv_w^T(192x576) + b, scatter to Q/K/Vt ----
  for (int nt = wave; nt < 36; nt += NHEADS) {
    int j = nt*16 + nn;                       // output column 0..575
    float bias = bf2f(gqkvB[j]);
    v8f acc[4];
#pragma unroll
    for (int mt = 0; mt < 4; ++mt)
#pragma unroll
      for (int v = 0; v < 8; ++v) acc[mt][v] = bias;
#pragma unroll
    for (int kc = 0; kc < 6; ++kc) {
      v16bf b = ld_frag(gqkvW + (size_t)j*CCH + kc*32, hf);
#pragma unroll
      for (int mt = 0; mt < 4; ++mt) {
        v16bf a = ld_frag(Xb + (mt*16 + nn)*XPITCH + kc*32, hf);
        acc[mt] = wmma_bf16(a, b, acc[mt]);
      }
    }
    int mi = nt / 12;                         // wave-uniform: 0=q 1=k 2=v
    int jr = (nt % 12)*16 + nn;
    int h = jr / HD, d = jr % HD;
    if (mi == 2) {
      // V stored transposed [h][d][r]: lane column d fixed, rows contiguous
      unsigned short* vbase = Vt + (h*32 + d)*64 + 8*hf;
#pragma unroll
      for (int mt = 0; mt < 4; ++mt)
#pragma unroll
        for (int v = 0; v < 8; v += 2)
          *(unsigned int*)(vbase + mt*16 + v) = f2bf_pk(acc[mt][v], acc[mt][v+1]);
    } else {
      unsigned short* base = (mi == 0 ? Qs : Ks) + (h*64 + 8*hf)*32 + d;
#pragma unroll
      for (int mt = 0; mt < 4; ++mt)
#pragma unroll
        for (int v = 0; v < 8; ++v)
          base[(mt*16 + v)*32] = f2bf(acc[mt][v]);
    }
  }
  __syncthreads();

  // ---- phase 2: per-head S = Q@K^T * scale, softmax -> Ps (wave == head) ----
  const int head = wave;
  {
    v8f S[4][4];
#pragma unroll
    for (int mt = 0; mt < 4; ++mt)
#pragma unroll
      for (int ct = 0; ct < 4; ++ct)
#pragma unroll
        for (int v = 0; v < 8; ++v) S[mt][ct][v] = 0.0f;

    v16bf aq[4];
#pragma unroll
    for (int mt = 0; mt < 4; ++mt)
      aq[mt] = ld_frag(Qs + (head*64 + mt*16 + nn)*32, hf);
#pragma unroll
    for (int ct = 0; ct < 4; ++ct) {
      v16bf bk = ld_frag(Ks + (head*64 + ct*16 + nn)*32, hf);
#pragma unroll
      for (int mt = 0; mt < 4; ++mt)
        S[mt][ct] = wmma_bf16(aq[mt], bk, S[mt][ct]);
    }
    const float scale = 0.20412414523193154f;  // 1/sqrt(24)
    const bool c3ok = (48 + nn) < WW;          // tile ct=3: only column 48 valid
#pragma unroll
    for (int mt = 0; mt < 4; ++mt) {
#pragma unroll
      for (int v = 0; v < 8; ++v) {
        float a0 = S[mt][0][v]*scale, a1 = S[mt][1][v]*scale;
        float a2 = S[mt][2][v]*scale, a3 = S[mt][3][v]*scale;
        float mx = fmaxf(fmaxf(a0, a1), fmaxf(a2, c3ok ? a3 : -3.0e38f));
#pragma unroll
        for (int sh = 1; sh < 16; sh <<= 1) mx = fmaxf(mx, __shfl_xor(mx, sh, 32));
        float e0 = __expf(a0 - mx), e1 = __expf(a1 - mx), e2 = __expf(a2 - mx);
        float e3 = c3ok ? __expf(a3 - mx) : 0.0f;
        float sum = e0 + e1 + e2 + e3;
#pragma unroll
        for (int sh = 1; sh < 16; sh <<= 1) sum += __shfl_xor(sum, sh, 32);
        float inv = __builtin_amdgcn_rcpf(sum);
        int r = mt*16 + v + 8*hf;
        unsigned short* pr = Ps + (head*64 + r)*64 + nn;
        pr[0]  = f2bf(e0*inv);
        pr[16] = f2bf(e1*inv);
        pr[32] = f2bf(e2*inv);
        pr[48] = f2bf(e3*inv);
      }
    }
  }

  // ---- phase 3: O = P(64x64) @ V(64x32) -> AO (concat heads, bf16) ----
  {
    v8f O[4][2];
#pragma unroll
    for (int mt = 0; mt < 4; ++mt)
#pragma unroll
      for (int dt = 0; dt < 2; ++dt)
#pragma unroll
        for (int v = 0; v < 8; ++v) O[mt][dt][v] = 0.0f;
#pragma unroll
    for (int kc = 0; kc < 2; ++kc) {
      v16bf bv0 = ld_frag(Vt + (head*32 + nn)*64 + kc*32, hf);
      v16bf bv1 = ld_frag(Vt + (head*32 + 16 + nn)*64 + kc*32, hf);
#pragma unroll
      for (int mt = 0; mt < 4; ++mt) {
        v16bf ap = ld_frag(Ps + (head*64 + mt*16 + nn)*64 + kc*32, hf);
        O[mt][0] = wmma_bf16(ap, bv0, O[mt][0]);
        O[mt][1] = wmma_bf16(ap, bv1, O[mt][1]);
      }
    }
#pragma unroll
    for (int dt = 0; dt < 2; ++dt) {
      int d = dt*16 + nn;
      if (d < HD) {
        int col = head*HD + d;
#pragma unroll
        for (int mt = 0; mt < 4; ++mt)
#pragma unroll
          for (int v = 0; v < 8; ++v) {
            int r = mt*16 + v + 8*hf;
            AO[r*XPITCH + col] = f2bf(O[mt][dt][v]);
          }
      }
    }
  }
  __syncthreads();

  // ---- phase 4: Y = AO(64x192) @ proj_w^T(192x192) + b -> OutS f32 ----
  for (int nt = wave; nt < 12; nt += NHEADS) {
    int col = nt*16 + nn;
    float bias = bf2f(gprojB[col]);
    v8f acc[4];
#pragma unroll
    for (int mt = 0; mt < 4; ++mt)
#pragma unroll
      for (int v = 0; v < 8; ++v) acc[mt][v] = bias;
#pragma unroll
    for (int kc = 0; kc < 6; ++kc) {
      v16bf b = ld_frag(gprojW + (size_t)col*CCH + kc*32, hf);
#pragma unroll
      for (int mt = 0; mt < 4; ++mt) {
        v16bf a = ld_frag(AO + (mt*16 + nn)*XPITCH + kc*32, hf);
        acc[mt] = wmma_bf16(a, b, acc[mt]);
      }
    }
#pragma unroll
    for (int mt = 0; mt < 4; ++mt)
#pragma unroll
      for (int v = 0; v < 8; ++v) {
        int r = mt*16 + v + 8*hf;
        if (r < WW) OutS[r*CCH + col] = acc[mt][v];
      }
  }
  __syncthreads();

  // ---- phase 5: window_reverse scatter to (N,C,H,W), w-contiguous runs ----
  for (int i = tid; i < CCH*WW; i += TPB) {
    int c = i / WW, p = i % WW;
    int wy = p / 7, wx = p % 7;
    out[(((size_t)nIdx*CCH + c)*56 + (whi*7 + wy))*56 + (wwi*7 + wx)] =
        OutS[p*CCH + c];
  }
}

extern "C" void kernel_launch(void* const* d_in, const int* in_sizes, int n_in,
                              void* d_out, int out_size, void* d_ws, size_t ws_size,
                              hipStream_t stream) {
  const float* x      = (const float*)d_in[0];
  const float* qkv_w  = (const float*)d_in[1];
  const float* qkv_b  = (const float*)d_in[2];
  const float* proj_w = (const float*)d_in[3];
  const float* proj_b = (const float*)d_in[4];
  unsigned short* ws  = (unsigned short*)d_ws;
  float* out          = (float*)d_out;

  (void)in_sizes; (void)n_in; (void)out_size; (void)ws_size;

  // bf16 weight conversion (tiny; weights stay hot in L2 across all windows)
  int prepBlocks = (576*192 + TPB - 1) / TPB;
  prep_weights<<<prepBlocks, TPB, 0, stream>>>(qkv_w, qkv_b, proj_w, proj_b, ws);

  // allow >64KB dynamic LDS (CDNA5 WGP has 320KB)
  hipFuncSetAttribute(reinterpret_cast<const void*>(win_mhsa),
                      hipFuncAttributeMaxDynamicSharedMemorySize, L_TOTAL);

  win_mhsa<<<4096, TPB, L_TOTAL, stream>>>(x, ws, out);
}
